// Attention_43963285242601
// MI455X (gfx1250) — matrix operand order repaired and checked by reading, not compiled, not run on hardware
//
#include <hip/hip_runtime.h>

// Problem constants (match reference)
#define B_    2
#define L_    2048
#define D_    4096
#define HQ    32
#define HKV   8
#define DH    128
#define GROUP 4                 // HQ / HKV
#define SCALE 0.08838834764831845f   // 1/sqrt(128)
#define THETA 500000.0f

typedef __attribute__((ext_vector_type(16))) _Float16 v16h;
typedef __attribute__((ext_vector_type(8)))  _Float16 v8h;
typedef __attribute__((ext_vector_type(8)))  float    v8f;

union Frag { v16h v; v8h h[2]; };

// ---- WMMA fragment loaders (CDNA5 16x16x32 f16 layouts) -------------------
// A (16x32, MxK): lane m=lane%16, kh=lane/16 holds K = kh*16+0..15 contiguous.
__device__ __forceinline__ v16h load_afrag(const _Float16* __restrict__ base, int ld) {
  int lane = threadIdx.x & 31;
  int m = lane & 15, kh = lane >> 4;
  const _Float16* p = base + (size_t)m * ld + kh * 16;
  Frag f;
  f.h[0] = *(const v8h*)(p);
  f.h[1] = *(const v8h*)(p + 8);
  return f.v;
}
// B (32x16, KxN): lane n=lane%16, kh=lane/16 holds K = kh*16+0..15 contiguous.
__device__ __forceinline__ v16h load_bfrag(const _Float16* __restrict__ base, int ld) {
  int lane = threadIdx.x & 31;
  int n = lane & 15, kh = lane >> 4;
  const _Float16* p = base + (size_t)n * ld + kh * 16;
  Frag f;
  f.h[0] = *(const v8h*)(p);
  f.h[1] = *(const v8h*)(p + 8);
  return f.v;
}

__device__ __forceinline__ v8f wmma_f16(v16h a, v16h b, v8f c) {
  return __builtin_amdgcn_wmma_f32_16x16x32_f16(false, a, false, b, (short)0, c, false, false);
}

// ---- CDNA5 async global->LDS copy (ASYNCcnt-tracked) -----------------------
// VDST VGPR holds the per-lane LDS byte address; VADDR holds the 64-bit global
// address (GV mode). Truncating a generic pointer to a __shared__ object gives
// the 32-bit LDS byte offset.
__device__ __forceinline__ void async_cp16(_Float16* lds_ptr, const _Float16* gptr) {
  unsigned lds = (unsigned)(unsigned long long)lds_ptr;
  asm volatile("global_load_async_to_lds_b128 %0, %1, off"
               :: "v"(lds), "v"(gptr) : "memory");
}
__device__ __forceinline__ void wait_async4() {
  asm volatile("s_wait_asynccnt 0x4" ::: "memory");
}
__device__ __forceinline__ void wait_async0() {
  asm volatile("s_wait_asynccnt 0x0" ::: "memory");
}

// ---- elementwise kernels ---------------------------------------------------
__global__ void cast_f32_to_f16(const float* __restrict__ in, _Float16* __restrict__ out, int n) {
  int i = blockIdx.x * blockDim.x + threadIdx.x;
  if (i < n) out[i] = (_Float16)in[i];
}

// In-place RoPE on (B*L, nh*128) f16, pos = row % L.
__global__ void rope_kernel(_Float16* __restrict__ t, int nh, int total) {
  int idx = blockIdx.x * blockDim.x + threadIdx.x;
  if (idx >= total) return;
  int i   = idx & 63;
  int h   = (idx >> 6) % nh;
  int row = idx / (64 * nh);
  int pos = row % L_;
  float inv = __powf(THETA, -(float)i / 64.0f);
  float ang = (float)pos * inv;
  float s, c;
  __sincosf(ang, &s, &c);
  size_t base = (size_t)row * nh * DH + h * DH + i;
  float t1 = (float)t[base];
  float t2 = (float)t[base + 64];
  t[base]      = (_Float16)(t1 * c - t2 * s);
  t[base + 64] = (_Float16)(t1 * s + t2 * c);
}

// V (B*L, HKV*DH) -> Vt (B, HKV, DH, L)
__global__ void transpose_v(const _Float16* __restrict__ V, _Float16* __restrict__ Vt, int n) {
  int idx = blockIdx.x * blockDim.x + threadIdx.x;
  if (idx >= n) return;
  int dh = idx & 127;
  int kv = (idx >> 7) & 7;
  int l  = (idx >> 10) % L_;
  int b  = idx / (1024 * L_);
  Vt[((size_t)((b * HKV + kv) * DH + dh)) * L_ + l] = V[idx];
}

// ---- GEMM: C[m,n] = sum_k A[m,k] * W[n,k]; A (M,K) f16, W (N,K) f16 -------
// Block = 256 threads = 8 waves (4 in M x 2 in N), wave tile 32x64, block
// tile 128x128. K staged through double-buffered LDS tiles filled with
// global_load_async_to_lds_b128 (ASYNCcnt), rows padded 32->40 halves.
#define TLD 40   // padded LDS row stride in halves

// Issue one 128x32 A tile + 128x32 B tile for k-block starting at k0.
// Each of the 256 threads issues 4 async b128 copies (2 A chunks + 2 B).
__device__ __forceinline__ void issue_tile(const _Float16* __restrict__ Ablk,
                                           const _Float16* __restrict__ Wblk,
                                           int K, int k0,
                                           _Float16* At, _Float16* Bt) {
  int t = threadIdx.x;
#pragma unroll
  for (int i = 0; i < 2; ++i) {
    int ch  = t + i * 256;        // 0..511 : 16-byte chunk id
    int row = ch >> 2;            // 0..127
    int c8  = (ch & 3) * 8;       // half offset within 32-wide row
    async_cp16(At + row * TLD + c8, Ablk + (size_t)row * K + k0 + c8);
    async_cp16(Bt + row * TLD + c8, Wblk + (size_t)row * K + k0 + c8);
  }
}

template <bool F32OUT>
__global__ __launch_bounds__(256) void gemm_nt(const _Float16* __restrict__ A,
                                               const _Float16* __restrict__ W,
                                               void* __restrict__ Cout,
                                               int M, int N, int K) {
  __shared__ __align__(16) _Float16 Atile[2][128][TLD];
  __shared__ __align__(16) _Float16 Btile[2][128][TLD];

  int wave = threadIdx.x >> 5;
  int wm = wave & 3, wn = wave >> 2;          // 4 x 2 wave grid
  int blockRow = blockIdx.y * 128;
  int blockCol = blockIdx.x * 128;
  const _Float16* Ablk = A + (size_t)blockRow * K;
  const _Float16* Wblk = W + (size_t)blockCol * K;

  v8f z = {};
  v8f acc[2][4];
#pragma unroll
  for (int mi = 0; mi < 2; ++mi)
#pragma unroll
    for (int nf = 0; nf < 4; ++nf) acc[mi][nf] = z;

  int nk = K / 32;
  issue_tile(Ablk, Wblk, K, 0, &Atile[0][0][0], &Btile[0][0][0]);

  for (int kb = 0; kb < nk; ++kb) {
    if (kb + 1 < nk) {
      int nb = (kb + 1) & 1;
      issue_tile(Ablk, Wblk, K, (kb + 1) * 32, &Atile[nb][0][0], &Btile[nb][0][0]);
      wait_async4();               // in-order: tile kb's 4 copies retired
    } else {
      wait_async0();
    }
    __syncthreads();               // all waves' copies visible

    int cb = kb & 1;
    const _Float16* At = &Atile[cb][0][0];
    const _Float16* Bt = &Btile[cb][0][0];
    // Batch all LDS fragment loads, then run the 8 WMMAs back-to-back.
    v16h a0 = load_afrag(At + (size_t)(wm * 32) * TLD, TLD);
    v16h a1 = load_afrag(At + (size_t)(wm * 32 + 16) * TLD, TLD);
    v16h b[4];
#pragma unroll
    for (int nf = 0; nf < 4; ++nf)
      b[nf] = load_bfrag(Bt + (size_t)(wn * 64 + nf * 16) * TLD, TLD);
#pragma unroll
    for (int nf = 0; nf < 4; ++nf) {
      acc[0][nf] = wmma_f16(a0, b[nf], acc[0][nf]);
      acc[1][nf] = wmma_f16(a1, b[nf], acc[1][nf]);
    }
    __syncthreads();               // done reading before buffer is refilled
  }

  int lane = threadIdx.x & 31, n = lane & 15, hl = lane >> 4;
#pragma unroll
  for (int mi = 0; mi < 2; ++mi)
#pragma unroll
    for (int nf = 0; nf < 4; ++nf)
#pragma unroll
      for (int r = 0; r < 8; ++r) {
        int row = blockRow + wm * 32 + mi * 16 + r + 8 * hl;
        int col = blockCol + wn * 64 + nf * 16 + n;
        float val = acc[mi][nf][r];
        if (F32OUT) ((float*)Cout)[(size_t)row * N + col] = val;
        else ((_Float16*)Cout)[(size_t)row * N + col] = (_Float16)val;
      }
}

// ---- Flash attention: one wave per (b, head, 16-row q tile) ----------------
__global__ __launch_bounds__(32) void attn_kernel(const _Float16* __restrict__ Q,
                                                  const _Float16* __restrict__ Kf,
                                                  const _Float16* __restrict__ Vt,
                                                  _Float16* __restrict__ Out) {
  __shared__ __align__(16) _Float16 Pt[16][32];
  int lane = threadIdx.x & 31, lcol = lane & 15, lhalf = lane >> 4;
  int qt = blockIdx.x, hq = blockIdx.y, b = blockIdx.z;
  int kv = hq / GROUP;
  int q0 = qt * 16;

  // Q fragments over the 4 K-chunks of head_dim=128
  v16h qa[4];
  const _Float16* qbase = Q + ((size_t)(b * L_ + q0)) * (HQ * DH) + hq * DH;
#pragma unroll
  for (int dc = 0; dc < 4; ++dc) qa[dc] = load_afrag(qbase + dc * 32, HQ * DH);

  v8f z = {};
  v8f o[8];
#pragma unroll
  for (int f = 0; f < 8; ++f) o[f] = z;
  float rmax[8], rsum[8];
#pragma unroll
  for (int r = 0; r < 8; ++r) { rmax[r] = -3.0e38f; rsum[r] = 0.0f; }

  int kbmax = (q0 + 15) / 32;
  for (int kb = 0; kb <= kbmax; ++kb) {
    int k0 = kb * 32;
#pragma unroll
    for (int nt = 0; nt < 2; ++nt) {
      int j0 = k0 + nt * 16;
      if (j0 <= q0 + 15) {
        v8f s = {};
        const _Float16* kbase = Kf + ((size_t)(b * L_ + j0)) * (HKV * DH) + kv * DH;
        // Batch the 4 K-fragment loads so they are all in flight before the
        // WMMA chain starts.
        v16h kfrag[4];
#pragma unroll
        for (int dc = 0; dc < 4; ++dc)
          kfrag[dc] = load_bfrag(kbase + dc * 32, HKV * DH);
#pragma unroll
        for (int dc = 0; dc < 4; ++dc)
          s = wmma_f16(qa[dc], kfrag[dc], s);
#pragma unroll
        for (int r = 0; r < 8; ++r) {
          float sv = s[r] * SCALE;
          int qrow = q0 + r + 8 * lhalf;
          int krow = j0 + lcol;
          if (krow > qrow) sv = -3.0e38f;      // causal mask
          float mx = sv;                        // half-wave row-max
          for (int off = 1; off < 16; off <<= 1) mx = fmaxf(mx, __shfl_xor(mx, off));
          float nm    = fmaxf(rmax[r], mx);
          float alpha = __expf(rmax[r] - nm);
          float p     = __expf(sv - nm);
          float ps    = p;                      // half-wave row-sum
          for (int off = 1; off < 16; off <<= 1) ps += __shfl_xor(ps, off);
          rsum[r] = rsum[r] * alpha + ps;
          rmax[r] = nm;
#pragma unroll
          for (int f = 0; f < 8; ++f) o[f][r] *= alpha;
          Pt[r + 8 * lhalf][nt * 16 + lcol] = (_Float16)p;
        }
      } else {
#pragma unroll
        for (int r = 0; r < 8; ++r) Pt[r + 8 * lhalf][nt * 16 + lcol] = (_Float16)0.0f;
      }
    }
    __syncthreads();
    // P as A-fragment (16x32) from LDS
    Frag pf;
    {
      const _Float16* pp = &Pt[lcol][lhalf * 16];
      pf.h[0] = *(const v8h*)(pp);
      pf.h[1] = *(const v8h*)(pp + 8);
    }
    // O += P (16x32) x V (32x128); Vt rows are contiguous in L.
    // Load all 8 V fragments first (keeps 16 b128 loads in flight), then run
    // the 8 independent WMMAs back-to-back.
    Frag vf[8];
#pragma unroll
    for (int f = 0; f < 8; ++f) {
      const _Float16* vb = Vt + ((size_t)((b * HKV + kv) * DH + f * 16 + lcol)) * L_ + k0 + lhalf * 16;
      vf[f].h[0] = *(const v8h*)(vb);
      vf[f].h[1] = *(const v8h*)(vb + 8);
    }
#pragma unroll
    for (int f = 0; f < 8; ++f)
      o[f] = wmma_f16(pf.v, vf[f].v, o[f]);
    __syncthreads();
  }

#pragma unroll
  for (int r = 0; r < 8; ++r) {
    float inv = 1.0f / rsum[r];
    int row = q0 + r + 8 * lhalf;
#pragma unroll
    for (int f = 0; f < 8; ++f)
      Out[((size_t)(b * L_ + row)) * (HQ * DH) + hq * DH + f * 16 + lcol] = (_Float16)(o[f][r] * inv);
  }
}

// ---- host launcher ---------------------------------------------------------
extern "C" void kernel_launch(void* const* d_in, const int* in_sizes, int n_in,
                              void* d_out, int out_size, void* d_ws, size_t ws_size,
                              hipStream_t stream) {
  const float* x  = (const float*)d_in[0];
  const float* Wq = (const float*)d_in[1];
  const float* Wk = (const float*)d_in[2];
  const float* Wv = (const float*)d_in[3];
  const float* Wo = (const float*)d_in[4];
  float* out = (float*)d_out;

  const size_t ML = (size_t)B_ * L_;           // 4096 rows
  char* ws = (char*)d_ws;
  size_t off = 0;
  auto take = [&](size_t elems) -> _Float16* {
    _Float16* p = (_Float16*)(ws + off);
    off += ((elems * sizeof(_Float16) + 255) & ~(size_t)255);
    return p;
  };

  _Float16* xh  = take(ML * D_);
  _Float16* Wqh = take((size_t)D_ * D_);
  _Float16* Wkh = take((size_t)HKV * DH * D_);
  _Float16* Wvh = take((size_t)HKV * DH * D_);
  _Float16* Woh = take((size_t)D_ * D_);
  _Float16* Qh  = take(ML * HQ * DH);
  _Float16* Kh  = take(ML * HKV * DH + 64 * (size_t)(HKV * DH)); // pad rows for masked frag loads
  _Float16* Vh  = take(ML * HKV * DH);
  _Float16* Vth = take(ML * HKV * DH + 4096);                    // pad tail
  _Float16* Ao  = take(ML * HQ * DH);

  auto cgrid = [](size_t n) { return dim3((unsigned)((n + 255) / 256)); };

  cast_f32_to_f16<<<cgrid(ML * D_), 256, 0, stream>>>(x,  xh,  (int)(ML * D_));
  cast_f32_to_f16<<<cgrid((size_t)D_ * D_), 256, 0, stream>>>(Wq, Wqh, D_ * D_);
  cast_f32_to_f16<<<cgrid((size_t)HKV * DH * D_), 256, 0, stream>>>(Wk, Wkh, HKV * DH * D_);
  cast_f32_to_f16<<<cgrid((size_t)HKV * DH * D_), 256, 0, stream>>>(Wv, Wvh, HKV * DH * D_);
  cast_f32_to_f16<<<cgrid((size_t)D_ * D_), 256, 0, stream>>>(Wo, Woh, D_ * D_);

  // Projections: M = B*L = 4096, K = D = 4096, block tile 128x128
  gemm_nt<false><<<dim3(D_ / 128, ML / 128), 256, 0, stream>>>(xh, Wqh, Qh, (int)ML, D_, D_);
  gemm_nt<false><<<dim3((HKV * DH) / 128, ML / 128), 256, 0, stream>>>(xh, Wkh, Kh, (int)ML, HKV * DH, D_);
  gemm_nt<false><<<dim3((HKV * DH) / 128, ML / 128), 256, 0, stream>>>(xh, Wvh, Vh, (int)ML, HKV * DH, D_);

  // RoPE in place
  rope_kernel<<<cgrid(ML * HQ * 64), 256, 0, stream>>>(Qh, HQ, (int)(ML * HQ * 64));
  rope_kernel<<<cgrid(ML * HKV * 64), 256, 0, stream>>>(Kh, HKV, (int)(ML * HKV * 64));

  // V -> Vt (b, kv, dh, L)
  transpose_v<<<cgrid(ML * HKV * DH), 256, 0, stream>>>(Vh, Vth, (int)(ML * HKV * DH));

  // Flash attention: one wave per 16 q-rows per head per batch
  attn_kernel<<<dim3(L_ / 16, HQ, B_), 32, 0, stream>>>(Qh, Kh, Vth, Ao);

  // Output projection, f32 store to d_out
  gemm_nt<true><<<dim3(D_ / 128, ML / 128), 256, 0, stream>>>(Ao, Woh, out, (int)ML, D_, D_);
}
